// Generator_78683800862785
// MI455X (gfx1250) — matrix operand order repaired and checked by reading, not compiled, not run on hardware
//
#include <hip/hip_runtime.h>
#include <hip/hip_bf16.h>
#include <math.h>

typedef __bf16 bf16;
typedef __attribute__((ext_vector_type(16))) __bf16 v16bf;
typedef __attribute__((ext_vector_type(8)))  __bf16 v8bf;
typedef __attribute__((ext_vector_type(8)))  float  v8f;

union ABfrag { v16bf v; v8bf h[2]; };

// ---------------------------------------------------------------------------
// Generic GEMM: C[M,N] = act(A[M,K] @ W[N,K]^T + bias[N])
// A: bf16 (or fp32 converted on load if AF32). W: bf16 row-major [N,K].
// One wave computes a 32(M) x 64(N) tile: 2 A-frags x 4 B-frags -> 8 accums
// of v_wmma_f32_16x16x32_bf16 (B loaded once, used twice => half W traffic).
// ACT: 0=none, 1=leaky-relu(0.2). OUTF32: write fp32 (LSTM gates) else bf16.
// Requires M % 32 == 0, N % 64 == 0, K % 32 == 0.
// ---------------------------------------------------------------------------
template<bool AF32, int ACT, bool OUTF32>
__global__ __launch_bounds__(256) void gemm_bf16(
    const void* __restrict__ Av, int lda,
    const bf16* __restrict__ W,
    const float* __restrict__ bias,
    void* __restrict__ Cv, int ldc,
    int M, int N, int K)
{
    const int tilesN = N >> 6;
    const int wid = blockIdx.x * 8 + (threadIdx.x >> 5);
    const int tm = wid / tilesN;
    const int tn = wid % tilesN;
    if (tm * 32 >= M) return;
    const int lane = threadIdx.x & 31;
    const int lm   = lane & 15;
    const bool hi  = lane >= 16;
    const int m0 = tm * 32;
    const int n0 = tn * 64;

    const bf16*  Ab = (const bf16*)Av;
    const float* Af = (const float*)Av;
    const long arow0 = (long)(m0 + lm) * (long)lda;
    const long arow1 = arow0 + 16L * lda;

    v8f acc[8] = {v8f{}, v8f{}, v8f{}, v8f{}, v8f{}, v8f{}, v8f{}, v8f{}};

    for (int k0 = 0; k0 < K; k0 += 32) {
        ABfrag a0, a1;
        const int ak = k0 + (hi ? 8 : 0);
        if (AF32) {
#pragma unroll
            for (int e = 0; e < 8; ++e) {
                a0.v[e]     = (bf16)Af[arow0 + ak + e];
                a0.v[e + 8] = (bf16)Af[arow0 + ak + 16 + e];
                a1.v[e]     = (bf16)Af[arow1 + ak + e];
                a1.v[e + 8] = (bf16)Af[arow1 + ak + 16 + e];
            }
        } else {
            a0.h[0] = *(const v8bf*)(Ab + arow0 + ak);
            a0.h[1] = *(const v8bf*)(Ab + arow0 + ak + 16);
            a1.h[0] = *(const v8bf*)(Ab + arow1 + ak);
            a1.h[1] = *(const v8bf*)(Ab + arow1 + ak + 16);
        }
        const int bk = k0 + (hi ? 8 : 0);
#pragma unroll
        for (int j = 0; j < 4; ++j) {
            ABfrag b;
            const bf16* wp = W + (long)(n0 + j * 16 + lm) * (long)K + bk;
            b.h[0] = *(const v8bf*)(wp);
            b.h[1] = *(const v8bf*)(wp + 16);
            acc[j] = __builtin_amdgcn_wmma_f32_16x16x32_bf16(
                false, a0.v, false, b.v, (short)0, acc[j], false, false);
            acc[4 + j] = __builtin_amdgcn_wmma_f32_16x16x32_bf16(
                false, a1.v, false, b.v, (short)0, acc[4 + j], false, false);
        }
    }

    float* Cf = (float*)Cv;
    bf16*  Cb = (bf16*)Cv;
#pragma unroll
    for (int j = 0; j < 4; ++j) {
        const int n = n0 + j * 16 + lm;
        const float bs = bias[n];
#pragma unroll
        for (int p = 0; p < 2; ++p) {
            long idx = (long)(m0 + 16 * p + (hi ? 8 : 0)) * ldc + n;
#pragma unroll
            for (int r = 0; r < 8; ++r) {
                float v = acc[4 * p + j][r] + bs;
                if (ACT == 1) v = (v >= 0.0f) ? v : 0.2f * v;
                if (OUTF32) Cf[idx] = v;
                else        Cb[idx] = (bf16)v;
                idx += ldc;
            }
        }
    }
}

// ---------------------------------------------------------------------------
// Single-query 4-head attention. One wave per (batch, head).
// q: [1024,512] bf16 (scale pre-folded). kv: [1024*100, 1024] bf16 (K | V).
// attout: [1024,512] bf16 (head-concat order).
// ---------------------------------------------------------------------------
__global__ __launch_bounds__(256) void attn_kernel(
    const bf16* __restrict__ q,
    const bf16* __restrict__ kv,
    bf16* __restrict__ attout)
{
    __shared__ float qs[8][128];
    const int w    = threadIdx.x >> 5;
    const int lane = threadIdx.x & 31;
    const int id = blockIdx.x * 8 + w;          // 0..4095
    const int b = id >> 2, h = id & 3;

    const bf16* qp = q + (long)b * 512 + h * 128;
#pragma unroll
    for (int i = 0; i < 4; ++i) qs[w][lane * 4 + i] = (float)qp[lane * 4 + i];

    const long rowbase = (long)b * 100;
    float sc[4] = {0.f, 0.f, 0.f, 0.f};
    for (int d0 = 0; d0 < 128; d0 += 8) {
        float qv[8];
#pragma unroll
        for (int e = 0; e < 8; ++e) qv[e] = qs[w][d0 + e];
#pragma unroll
        for (int j = 0; j < 4; ++j) {
            const int s = lane + 32 * j;
            if (s < 100) {
                const bf16* kp = kv + (rowbase + s) * 1024 + h * 128 + d0;
#pragma unroll
                for (int e = 0; e < 8; ++e) sc[j] += qv[e] * (float)kp[e];
            }
        }
    }
    // softmax over S=100 (lane owns s = lane + 32j)
    float mx = -1e30f;
#pragma unroll
    for (int j = 0; j < 4; ++j) if (lane + 32 * j < 100) mx = fmaxf(mx, sc[j]);
    for (int o = 16; o; o >>= 1) mx = fmaxf(mx, __shfl_xor(mx, o, 32));
    float ex[4]; float sum = 0.f;
#pragma unroll
    for (int j = 0; j < 4; ++j) {
        ex[j] = (lane + 32 * j < 100) ? __expf(sc[j] - mx) : 0.0f;
        sum += ex[j];
    }
    for (int o = 16; o; o >>= 1) sum += __shfl_xor(sum, o, 32);
    const float inv = 1.0f / sum;
#pragma unroll
    for (int j = 0; j < 4; ++j) ex[j] *= inv;

    // att[d] = sum_s w[s] * V[s,d]; lane owns d = 4*lane..4*lane+3
    float acc[4] = {0.f, 0.f, 0.f, 0.f};
#pragma unroll
    for (int j = 0; j < 4; ++j) {
        const int smax = (j == 3) ? 4 : 32;
        for (int si = 0; si < smax; ++si) {
            const float wv = __shfl(ex[j], si, 32);
            const int s = 32 * j + si;
            const bf16* vp = kv + (rowbase + s) * 1024 + 512 + h * 128 + lane * 4;
#pragma unroll
            for (int i = 0; i < 4; ++i) acc[i] += wv * (float)vp[i];
        }
    }
    bf16* op = attout + (long)b * 512 + h * 128 + lane * 4;
#pragma unroll
    for (int i = 0; i < 4; ++i) op[i] = (bf16)acc[i];
}

// ---------------------------------------------------------------------------
// LSTM pointwise update. gates fp32 [1024,2048] (i|f|g|o). c fp32 [1024,512].
// Writes h (bf16) into two destinations (pre-offset pointers).
// ---------------------------------------------------------------------------
__global__ __launch_bounds__(256) void lstm_update(
    const float* __restrict__ g, float* __restrict__ c,
    bf16* __restrict__ d0, int ld0, bf16* __restrict__ d1, int ld1)
{
    const int idx = blockIdx.x * 256 + threadIdx.x;   // 0..524287
    const int b = idx >> 9, j = idx & 511;
    const float* gr = g + (long)b * 2048;
    const float gi = gr[j], gf = gr[512 + j], gg = gr[1024 + j], go = gr[1536 + j];
    const float si = 1.0f / (1.0f + __expf(-gi));
    const float sf = 1.0f / (1.0f + __expf(-gf));
    const float so = 1.0f / (1.0f + __expf(-go));
    const float c2 = sf * c[idx] + si * tanhf(gg);
    const float hh = so * tanhf(c2);
    c[idx] = c2;
    d0[(long)b * ld0 + j] = (bf16)hh;
    d1[(long)b * ld1 + j] = (bf16)hh;
}

// fc2c: [1024,128] @ [2,128]^T + b, tanh -> d_out[b, t, 0..1] fp32
__global__ __launch_bounds__(256) void fc2c_kernel(
    const bf16* __restrict__ x, const bf16* __restrict__ w,
    const float* __restrict__ bias, float* __restrict__ out, int t)
{
    const int idx = blockIdx.x * 256 + threadIdx.x;   // 0..2047
    const int b = idx >> 1, o = idx & 1;
    float s = bias[o];
    const bf16* xp = x + (long)b * 128;
    const bf16* wp = w + o * 128;
    for (int k = 0; k < 128; ++k) s += (float)xp[k] * (float)wp[k];
    out[((long)b * 30 + t) * 2 + o] = tanhf(s);
}

// zt -> xcat[:, 512:576] bf16
__global__ __launch_bounds__(256) void zcopy(
    bf16* __restrict__ xcat, const float* __restrict__ z, int t)
{
    const int idx = blockIdx.x * 256 + threadIdx.x;   // 0..65535
    const int b = idx >> 6, l = idx & 63;
    xcat[(long)b * 576 + 512 + l] = (bf16)z[((long)b * 30 + t) * 64 + l];
}

// fp32 -> bf16 (with optional scale)
__global__ __launch_bounds__(256) void cvt_f32_bf16(
    bf16* __restrict__ dst, const float* __restrict__ src, int n, float scale)
{
    const int i = blockIdx.x * 256 + threadIdx.x;
    if (i < n) dst[i] = (bf16)(src[i] * scale);
}

// fused bias prep: dst = (a + b?) * scale, fp32
__global__ __launch_bounds__(256) void bias_prep(
    float* __restrict__ dst, const float* __restrict__ a,
    const float* __restrict__ b, int n, float scale)
{
    const int i = blockIdx.x * 256 + threadIdx.x;
    if (i < n) dst[i] = (a[i] + (b ? b[i] : 0.0f)) * scale;
}

// fuse LSTM weights: dst[2048,1024] = [wih | whh] (per-row concat), bf16
__global__ __launch_bounds__(256) void lstm_fuse_w(
    bf16* __restrict__ dst, const float* __restrict__ wih,
    const float* __restrict__ whh)
{
    const int i = blockIdx.x * 256 + threadIdx.x;
    if (i >= 2048 * 1024) return;
    const int r = i >> 10, col = i & 1023;
    const float v = (col < 512) ? wih[r * 512 + col] : whh[r * 512 + (col - 512)];
    dst[i] = (bf16)v;
}

__global__ __launch_bounds__(256) void zero_ws(unsigned int* __restrict__ p, long n)
{
    long i = (long)blockIdx.x * 256 + threadIdx.x;
    const long stride = (long)gridDim.x * 256;
    for (; i < n; i += stride) p[i] = 0u;
}

// ---------------------------------------------------------------------------
extern "C" void kernel_launch(void* const* d_in, const int* in_sizes, int n_in,
                              void* d_out, int out_size, void* d_ws, size_t ws_size,
                              hipStream_t stream)
{
    (void)in_sizes; (void)n_in; (void)out_size; (void)ws_size;
    const float* encoded = (const float*)d_in[0];
    const float* z       = (const float*)d_in[1];
    const float* q_w     = (const float*)d_in[2];
    const float* k_w     = (const float*)d_in[3];
    const float* v_w     = (const float*)d_in[4];
    const float* q_b     = (const float*)d_in[5];
    const float* k_b     = (const float*)d_in[6];
    const float* v_b     = (const float*)d_in[7];
    const float* out_w   = (const float*)d_in[8];
    const float* out_b   = (const float*)d_in[9];
    const float* fc1_w   = (const float*)d_in[10];
    const float* fc1_b   = (const float*)d_in[11];
    const float* l1_wih  = (const float*)d_in[12];
    const float* l1_whh  = (const float*)d_in[13];
    const float* l1_bih  = (const float*)d_in[14];
    const float* l1_bhh  = (const float*)d_in[15];
    const float* l2_wih  = (const float*)d_in[16];
    const float* l2_whh  = (const float*)d_in[17];
    const float* l2_bih  = (const float*)d_in[18];
    const float* l2_bhh  = (const float*)d_in[19];
    const float* fc2a_w  = (const float*)d_in[20];
    const float* fc2a_b  = (const float*)d_in[21];
    const float* fc2b_w  = (const float*)d_in[22];
    const float* fc2b_b  = (const float*)d_in[23];
    const float* fc2c_w  = (const float*)d_in[24];
    const float* fc2c_b  = (const float*)d_in[25];

    char* ws = (char*)d_ws;
    size_t off = 0;
    auto al = [&](size_t bytes) -> char* {
        off = (off + 255) & ~(size_t)255;
        char* p = ws + off; off += bytes; return p;
    };

    // bf16 weights
    bf16* kvw  = (bf16*)al((size_t)1024 * 256 * 2);  // [k_w ; v_w]
    bf16* qw   = (bf16*)al((size_t)512 * 512 * 2);   // pre-scaled
    bf16* outw = (bf16*)al((size_t)512 * 512 * 2);
    bf16* fc1w = (bf16*)al((size_t)512 * 576 * 2);
    bf16* l1w  = (bf16*)al((size_t)2048 * 1024 * 2); // [wih | whh]
    bf16* l2w  = (bf16*)al((size_t)2048 * 1024 * 2);
    bf16* f2aw = (bf16*)al((size_t)256 * 512 * 2);
    bf16* f2bw = (bf16*)al((size_t)128 * 256 * 2);
    bf16* f2cw = (bf16*)al((size_t)2 * 128 * 2);
    // fp32 fused biases
    float* kvb = (float*)al((size_t)1024 * 4);
    float* qb  = (float*)al((size_t)512 * 4);
    float* l1b = (float*)al((size_t)2048 * 4);
    float* l2b = (float*)al((size_t)2048 * 4);
    // K|V cache, bf16: row = b*100+s, cols 0..511 = K heads, 512..1023 = V
    bf16* kvout = (bf16*)al((size_t)102400 * 1024 * 2);
    // zero-initialized recurrent state (contiguous region)
    size_t state_begin = (off + 255) & ~(size_t)255;
    float* c1    = (float*)al((size_t)1024 * 512 * 4);
    float* c2    = (float*)al((size_t)1024 * 512 * 4);
    bf16*  xh1   = (bf16*)al((size_t)1024 * 1024 * 2); // [fc1_out | h1]
    bf16*  xh2   = (bf16*)al((size_t)1024 * 1024 * 2); // [h1 | h2]
    bf16*  attin = (bf16*)al((size_t)1024 * 512 * 2);  // h2 (q-proj / fc2a input)
    size_t state_end = off;
    // per-step scratch
    bf16*  qbuf  = (bf16*)al((size_t)1024 * 512 * 2);
    bf16*  atth  = (bf16*)al((size_t)1024 * 512 * 2);
    bf16*  xcat  = (bf16*)al((size_t)1024 * 576 * 2);  // [att | zt]
    float* gates = (float*)al((size_t)1024 * 2048 * 4);
    bf16*  f2ao  = (bf16*)al((size_t)1024 * 256 * 2);
    bf16*  f2bo  = (bf16*)al((size_t)1024 * 128 * 2);

    const float qscale = 0.0883883476483184f; // 1/sqrt(128), folded into q_w/q_b

    // ---- weight prep (cheap, once per call) ----
    cvt_f32_bf16<<<(131072 + 255) / 256, 256, 0, stream>>>(kvw, k_w, 131072, 1.0f);
    cvt_f32_bf16<<<(131072 + 255) / 256, 256, 0, stream>>>(kvw + 131072, v_w, 131072, 1.0f);
    cvt_f32_bf16<<<(262144 + 255) / 256, 256, 0, stream>>>(qw, q_w, 262144, qscale);
    cvt_f32_bf16<<<(262144 + 255) / 256, 256, 0, stream>>>(outw, out_w, 262144, 1.0f);
    cvt_f32_bf16<<<(294912 + 255) / 256, 256, 0, stream>>>(fc1w, fc1_w, 294912, 1.0f);
    lstm_fuse_w<<<(2048 * 1024 + 255) / 256, 256, 0, stream>>>(l1w, l1_wih, l1_whh);
    lstm_fuse_w<<<(2048 * 1024 + 255) / 256, 256, 0, stream>>>(l2w, l2_wih, l2_whh);
    cvt_f32_bf16<<<(131072 + 255) / 256, 256, 0, stream>>>(f2aw, fc2a_w, 131072, 1.0f);
    cvt_f32_bf16<<<(32768 + 255) / 256, 256, 0, stream>>>(f2bw, fc2b_w, 32768, 1.0f);
    cvt_f32_bf16<<<1, 256, 0, stream>>>(f2cw, fc2c_w, 256, 1.0f);
    bias_prep<<<2, 256, 0, stream>>>(kvb, k_b, nullptr, 512, 1.0f);
    bias_prep<<<2, 256, 0, stream>>>(kvb + 512, v_b, nullptr, 512, 1.0f);
    bias_prep<<<2, 256, 0, stream>>>(qb, q_b, nullptr, 512, qscale);
    bias_prep<<<8, 256, 0, stream>>>(l1b, l1_bih, l1_bhh, 2048, 1.0f);
    bias_prep<<<8, 256, 0, stream>>>(l2b, l2_bih, l2_bhh, 2048, 1.0f);

    // ---- zero recurrent state every call (deterministic) ----
    const long state_words = (long)(state_end - state_begin) / 4;
    zero_ws<<<2048, 256, 0, stream>>>((unsigned int*)(ws + state_begin), state_words);

    // ---- K/V projection: [102400,256] x [1024,256]^T -> kvout ----
    {
        const int M = 102400, N = 1024, K = 256;
        const int blocks = (M / 32) * (N / 64) / 8; // 6400
        gemm_bf16<true, 0, false><<<blocks, 256, 0, stream>>>(
            encoded, 256, kvw, kvb, kvout, 1024, M, N, K);
    }

    // ---- 30 sequential decode steps ----
    for (int t = 0; t < 30; ++t) {
        // q = (h2 @ q_w^T + q_b) * scale   (scale folded)
        gemm_bf16<false, 0, false><<<32, 256, 0, stream>>>(
            attin, 512, qw, qb, qbuf, 512, 1024, 512, 512);
        // attention -> atth [1024,512]
        attn_kernel<<<512, 256, 0, stream>>>(qbuf, kvout, atth);
        // out proj -> xcat cols [0,512)
        gemm_bf16<false, 0, false><<<32, 256, 0, stream>>>(
            atth, 512, outw, out_b, xcat, 576, 1024, 512, 512);
        // z_t -> xcat cols [512,576)
        zcopy<<<256, 256, 0, stream>>>(xcat, z, t);
        // fc1 + lrelu -> xh1 cols [0,512)
        gemm_bf16<false, 1, false><<<32, 256, 0, stream>>>(
            xcat, 576, fc1w, fc1_b, xh1, 1024, 1024, 512, 576);
        // LSTM1 gates -> fp32, then pointwise update (h1 -> xh2[:,0:512], xh1[:,512:])
        gemm_bf16<false, 0, true><<<128, 256, 0, stream>>>(
            xh1, 1024, l1w, l1b, gates, 2048, 1024, 2048, 1024);
        lstm_update<<<2048, 256, 0, stream>>>(gates, c1, xh2, 1024, xh1 + 512, 1024);
        // LSTM2 gates + update (h2 -> xh2[:,512:], attin)
        gemm_bf16<false, 0, true><<<128, 256, 0, stream>>>(
            xh2, 1024, l2w, l2b, gates, 2048, 1024, 2048, 1024);
        lstm_update<<<2048, 256, 0, stream>>>(gates, c2, xh2 + 512, 1024, attin, 512);
        // fc2a + lrelu, fc2b + lrelu
        gemm_bf16<false, 1, false><<<16, 256, 0, stream>>>(
            attin, 512, f2aw, fc2a_b, f2ao, 256, 1024, 256, 512);
        gemm_bf16<false, 1, false><<<8, 256, 0, stream>>>(
            f2ao, 256, f2bw, fc2b_b, f2bo, 128, 1024, 128, 256);
        // fc2c + tanh -> d_out[:, t, :]
        fc2c_kernel<<<8, 256, 0, stream>>>(f2bo, f2cw, fc2c_b, (float*)d_out, t);
    }
}
